// Net_12902081757308
// MI455X (gfx1250) — matrix operand order repaired and checked by reading, not compiled, not run on hardware
//
#include <hip/hip_runtime.h>
#include <hip/hip_bf16.h>

// ---------------------------------------------------------------------------
// MoE 3D conv (E=5, Ci=32, Co=64, K=5) + InstanceNorm + Mish for gfx1250.
// Conv as implicit GEMM on v_wmma_f32_16x16x32_f16:
//   A = per-sample mixed weights [64 x 4000] f16, pre-packed in WMMA lane order,
//       staged per-tap (4 KB) into LDS via global_load_async_to_lds_b128 with
//       double buffering (ASYNCcnt + workgroup barriers).
//   B = channels-last input patches [32 x 16 positions] f16, one contiguous
//       32B load per lane (zeroed at halos so EXEC stays all-ones at WMMA).
//   C = f32 accumulators; each wave owns 2 position-tiles x all 64 Co
//       -> 8 WMMAs per tap per wave.
// Workspace (~143.7 MB):
//   [0)           gates  g[4*5*64] f32
//   [8192)        packed A apk[4*125*4*32*16] f16           (2,048,000 B)
//   [2,056,192)   xT[4*110592*32] f16 channels-last         (28,311,552 B)
//   [30,367,744)  y[4*64*110592] f32 pre-norm conv output   (113,246,208 B)
//   [143,613,952) stats[4*64*2] f32 (mu, rsigma)
// ---------------------------------------------------------------------------

typedef _Float16 v16h __attribute__((ext_vector_type(16)));
typedef float    v8f  __attribute__((ext_vector_type(8)));

#define NS    4
#define CI_   32
#define CO_   64
#define NE    5
#define TAPS  125
#define D48   48
#define POSN  110592          // 48^3
#define TILES_PER_N 6912      // POSN / 16

// ---------------------------------------------------------------------------
// 1) Gates: g[n, e, co] = softmax_e( t[n,:] . gate_w[e*64+co,:] + gate_b )
// ---------------------------------------------------------------------------
__global__ __launch_bounds__(256) void gates_kernel(
    const float* __restrict__ t, const float* __restrict__ gw,
    const float* __restrict__ gb, float* __restrict__ g) {
  const int idx = threadIdx.x;          // 0..255 = n*64 + co
  const int n = idx >> 6, co = idx & 63;
  const float t0 = t[n * 3 + 0], t1 = t[n * 3 + 1], t2 = t[n * 3 + 2];
  float l[NE];
  #pragma unroll
  for (int e = 0; e < NE; ++e) {
    const int row = e * CO_ + co;
    l[e] = t0 * gw[row * 3 + 0] + t1 * gw[row * 3 + 1] + t2 * gw[row * 3 + 2] + gb[row];
  }
  float m = l[0];
  #pragma unroll
  for (int e = 1; e < NE; ++e) m = fmaxf(m, l[e]);
  float s = 0.f;
  #pragma unroll
  for (int e = 0; e < NE; ++e) { l[e] = expf(l[e] - m); s += l[e]; }
  const float inv = 1.0f / s;
  #pragma unroll
  for (int e = 0; e < NE; ++e) g[(n * NE + e) * CO_ + co] = l[e] * inv;
}

// ---------------------------------------------------------------------------
// 2) Pack mixed weights into WMMA A-fragment order (f16).
//    apk[((n*125 + tap)*4 + mt)*32 + lane][16 halves]
//    Lane l holds M = mt*16 + (l%16); element e holds K(=ci):
//      ci = 16*(l/16) + e     (contiguous order, matches B)
// ---------------------------------------------------------------------------
__global__ __launch_bounds__(256) void pack_kernel(
    const float* __restrict__ g,  const float* __restrict__ w5,
    const float* __restrict__ w3, const float* __restrict__ w1,
    const float* __restrict__ wa3, const float* __restrict__ wa5,
    _Float16* __restrict__ apk) {
  const int idx = blockIdx.x * 256 + threadIdx.x;   // 0..63999
  const int lane = idx & 31;
  const int mt   = (idx >> 5) & 3;
  const int tap  = (idx >> 7) % TAPS;
  const int n    = idx / (128 * TAPS);
  const int co   = mt * 16 + (lane & 15);
  const int c8   = (lane >> 4) * 8;
  const int kd = tap / 25, kh = (tap / 5) % 5, kw = tap % 5;
  const bool in3 = (kd >= 1 && kd <= 3) && (kh >= 1 && kh <= 3) && (kw >= 1 && kw <= 3);
  const bool ctr = (tap == 62);   // (2,2,2)
  const float* gp = g + n * (NE * CO_) + co;
  const float g0 = gp[0], g1 = gp[64], g2 = gp[128], g3 = gp[192], g4 = gp[256];
  v16h o;
  #pragma unroll
  for (int e = 0; e < 16; ++e) {
    const int ci = 16 * (lane >> 4) + e;
    const int wi = co * CI_ + ci;
    float v = g0 * w5[wi * TAPS + tap] + g4 * wa5[wi] * (1.0f / 125.0f);
    if (in3) v += g1 * w3[wi * 27 + (kd - 1) * 9 + (kh - 1) * 3 + (kw - 1)]
                + g3 * wa3[wi] * (1.0f / 27.0f);
    if (ctr) v += g2 * w1[wi];
    o[e] = (_Float16)v;
  }
  *(v16h*)(apk + (size_t)idx * 16) = o;
}

// ---------------------------------------------------------------------------
// 3) Transpose x [n, ci, pos] f32 -> xT [n, pos, ci] f16 (channels-last).
// ---------------------------------------------------------------------------
__global__ __launch_bounds__(256) void transpose_kernel(
    const float* __restrict__ x, _Float16* __restrict__ xT) {
  const int idx = blockIdx.x * 256 + threadIdx.x;   // n*POSN + pos
  const int n = idx / POSN, pos = idx % POSN;
  const float* xp = x + (size_t)n * CI_ * POSN + pos;
  v16h a, b;
  #pragma unroll
  for (int c = 0; c < 16; ++c) a[c] = (_Float16)xp[(size_t)c * POSN];
  #pragma unroll
  for (int c = 0; c < 16; ++c) b[c] = (_Float16)xp[(size_t)(16 + c) * POSN];
  v16h* o = (v16h*)(xT + (size_t)idx * CI_);
  o[0] = a;
  o[1] = b;
}

// ---------------------------------------------------------------------------
// 4) WMMA conv. Block = 8 waves, one sample; per-tap A slice (4 KB) staged in
//    LDS by async copies, double buffered. Wave = 2 position-tiles x 64 Co.
// ---------------------------------------------------------------------------
__global__ __launch_bounds__(256) void conv_wmma_kernel(
    const _Float16* __restrict__ xT, const _Float16* __restrict__ apk,
    float* __restrict__ y) {
  __shared__ __align__(64) _Float16 abuf[2][2048];   // 2 x 4KB tap slices

  const int tid  = threadIdx.x;
  const int lane = tid & 31;
  const int gwave = (blockIdx.x * 256 + tid) >> 5;   // 0..13823
  const int n  = gwave / (TILES_PER_N / 2);
  const int tp = gwave % (TILES_PER_N / 2);
  const int nl = lane & 15;                // N index (position in tile)
  const int hl = lane >> 4;                // lane half

  int rowp[2], w0p[2], dp[2], hp[2];
  #pragma unroll
  for (int s = 0; s < 2; ++s) {
    const int tl = tp * 2 + s;
    rowp[s] = tl / 3;                      // d*48 + h
    w0p[s]  = (tl % 3) * 16;
    dp[s]   = rowp[s] / D48;
    hp[s]   = rowp[s] % D48;
  }

  const _Float16* xn  = xT + (size_t)n * POSN * CI_;
  const _Float16* apn = apk + (size_t)n * TAPS * 2048;   // 2048 halves per tap
  const unsigned ldsbase = (unsigned)(uintptr_t)(&abuf[0][0]);

  // Each of the 256 threads async-copies 16B: 4KB per tap slice per block.
  auto issueA = [&](int tap, int bufi) {
    const unsigned dst = ldsbase + (unsigned)bufi * 4096u + (unsigned)tid * 16u;
    const _Float16* src = apn + (size_t)tap * 2048 + tid * 8;
    asm volatile("global_load_async_to_lds_b128 %0, %1, off"
                 :: "v"(dst), "v"(src) : "memory");
  };

  v8f acc[2][4] = {};

  issueA(0, 0);
  issueA(1, 1);

  int tap = 0;
  for (int kd = 0; kd < 5; ++kd) {
    for (int kh = 0; kh < 5; ++kh) {
      #pragma unroll
      for (int kw = 0; kw < 5; ++kw, ++tap) {
        // Async loads complete in order: <=1 outstanding means tap's slice is
        // in LDS (for the last tap wait for 0). Barrier covers other waves.
        if (tap == TAPS - 1) asm volatile("s_wait_asynccnt 0x0" ::: "memory");
        else                 asm volatile("s_wait_asynccnt 0x1" ::: "memory");
        __syncthreads();

        v16h b[2] = {};
        #pragma unroll
        for (int s = 0; s < 2; ++s) {
          const int dd = dp[s] + kd - 2;
          const int hh = hp[s] + kh - 2;
          const int ww = w0p[s] + nl + kw - 2;
          if (((unsigned)dd < 48u) & ((unsigned)hh < 48u) & ((unsigned)ww < 48u))
            b[s] = *(const v16h*)(xn +
                     (size_t)(dd * 2304 + hh * D48 + ww) * CI_ + 16 * hl);
        }

        const v16h* aT = (const v16h*)(abuf[tap & 1]);
        #pragma unroll
        for (int m = 0; m < 4; ++m) {
          const v16h a = aT[m * 32 + lane];
          #pragma unroll
          for (int s = 0; s < 2; ++s)
            acc[s][m] = __builtin_amdgcn_wmma_f32_16x16x32_f16(
                /*neg_a=*/false, a, /*neg_b=*/false, b[s],
                /*c_mod=*/(short)0, acc[s][m],
                /*reuse_a=*/false, /*reuse_b=*/false);
        }

        __syncthreads();   // all waves done reading buf[tap&1]
        if (tap + 2 < TAPS) issueA(tap + 2, tap & 1);
      }
    }
  }

  // C layout: element v of lane l -> M = v + 8*(l/16), N = l%16
  #pragma unroll
  for (int s = 0; s < 2; ++s) {
    float* yp = y + (size_t)n * CO_ * POSN + (size_t)rowp[s] * D48 + w0p[s] + nl;
    #pragma unroll
    for (int m = 0; m < 4; ++m) {
      #pragma unroll
      for (int v = 0; v < 8; ++v) {
        const int co = m * 16 + hl * 8 + v;
        yp[(size_t)co * POSN] = acc[s][m][v];
      }
    }
  }
}

// ---------------------------------------------------------------------------
// 5) Per-(n,co) mean / rsigma over 48^3 positions.
// ---------------------------------------------------------------------------
__global__ __launch_bounds__(256) void stats_kernel(
    const float* __restrict__ y, float* __restrict__ stats) {
  const int b = blockIdx.x;                 // n*64 + co
  const float* yp = y + (size_t)b * POSN;
  float s = 0.f, s2 = 0.f;
  for (int i = threadIdx.x; i < POSN; i += 256) {
    const float v = yp[i];
    s += v;
    s2 += v * v;
  }
  __shared__ float sh[256], sh2[256];
  sh[threadIdx.x] = s;
  sh2[threadIdx.x] = s2;
  __syncthreads();
  for (int o = 128; o > 0; o >>= 1) {
    if (threadIdx.x < o) {
      sh[threadIdx.x]  += sh[threadIdx.x + o];
      sh2[threadIdx.x] += sh2[threadIdx.x + o];
    }
    __syncthreads();
  }
  if (threadIdx.x == 0) {
    const float inv = 1.0f / (float)POSN;
    const float mu  = sh[0] * inv;
    const float var = fmaxf(sh2[0] * inv - mu * mu, 0.0f);
    stats[2 * b]     = mu;
    stats[2 * b + 1] = rsqrtf(var + 1e-5f);
  }
}

// ---------------------------------------------------------------------------
// 6) InstanceNorm (affine) + Mish.
// ---------------------------------------------------------------------------
__global__ __launch_bounds__(256) void norm_mish_kernel(
    const float* __restrict__ y, const float* __restrict__ stats,
    const float* __restrict__ gamma, const float* __restrict__ beta,
    float* __restrict__ out) {
  const size_t idx = (size_t)blockIdx.x * 256 + threadIdx.x;
  const int nc = (int)(idx / POSN);         // n*64 + co
  const int co = nc & 63;
  const float mu = stats[2 * nc], rs = stats[2 * nc + 1];
  const float z = (y[idx] - mu) * rs * gamma[co] + beta[co];
  const float sp = (z > 20.0f) ? z : log1pf(expf(z));
  out[idx] = z * tanhf(sp);
}

// ---------------------------------------------------------------------------
extern "C" void kernel_launch(void* const* d_in, const int* in_sizes, int n_in,
                              void* d_out, int out_size, void* d_ws, size_t ws_size,
                              hipStream_t stream) {
  const float* x     = (const float*)d_in[0];
  const float* t     = (const float*)d_in[1];
  const float* w5    = (const float*)d_in[2];
  const float* w3    = (const float*)d_in[3];
  const float* w1    = (const float*)d_in[4];
  const float* wa3   = (const float*)d_in[5];
  const float* wa5   = (const float*)d_in[6];
  const float* gw    = (const float*)d_in[7];
  const float* gb    = (const float*)d_in[8];
  const float* gamma = (const float*)d_in[9];
  const float* beta  = (const float*)d_in[10];
  float* out = (float*)d_out;

  char* ws = (char*)d_ws;
  float*    g_gate = (float*)(ws);
  _Float16* apk    = (_Float16*)(ws + 8192);
  _Float16* xT     = (_Float16*)(ws + 8192 + 2048000);
  float*    y      = (float*)(ws + 8192 + 2048000 + 28311552);
  float*    stats  = (float*)(ws + 8192 + 2048000 + 28311552 + 113246208);

  gates_kernel<<<1, 256, 0, stream>>>(t, gw, gb, g_gate);
  pack_kernel<<<250, 256, 0, stream>>>(g_gate, w5, w3, w1, wa3, wa5, apk);
  transpose_kernel<<<1728, 256, 0, stream>>>(x, xT);
  conv_wmma_kernel<<<1728, 256, 0, stream>>>(xT, apk, y);
  stats_kernel<<<NS * CO_, 256, 0, stream>>>(y, stats);
  norm_mish_kernel<<<POSN, 256, 0, stream>>>(y, stats, gamma, beta, out);
}
